// OracleLocalDeltaRewriteModel_43696997269506
// MI455X (gfx1250) — matrix operand order repaired and checked by reading, not compiled, not run on hardware
//
#include <hip/hip_runtime.h>

// ---------------------------------------------------------------------------
// Problem constants (match the JAX reference)
// ---------------------------------------------------------------------------
constexpr int   B_  = 2;
constexpr int   N_  = 512;
constexpr int   H_  = 128;
constexpr int   FD_ = 5;
constexpr int   NT_ = 3;
constexpr int   SD_ = 4;
constexpr float COPYV = 10.0f;
constexpr float EPS_  = 1e-6f;

typedef __attribute__((ext_vector_type(16))) _Float16     v16h;
typedef __attribute__((ext_vector_type(8)))  float        v8f;
typedef __attribute__((ext_vector_type(8)))  unsigned int v8u;
typedef __attribute__((ext_vector_type(4)))  unsigned int u32x4;
typedef __attribute__((ext_vector_type(8)))  int          i32x8;
typedef __attribute__((ext_vector_type(4)))  int          i32x4;

// D = A(16x32 f16) * B(32x16 f16) + C(16x16 f32)
#define WMMA_F16(A, Bm, C) \
  __builtin_amdgcn_wmma_f32_16x16x32_f16(false, (A), false, (Bm), (short)0, (C), false, false)

// 16-bit A-matrix 16x32, contiguous k order used here (k = 16*half + 2r): lane = M row; VGPR r holds K pair.
__device__ __forceinline__ int kA16(int r, int half) {
  return half * 16 + 2 * r;
}

// Fragment-linear index for a (K x 128) f16 B operand:
// element (k, n) lives at ((kc*8+nt)*32 + lane)*16 + slot with
// kc=k/32, half=(k%32)/16, slot=k%16, lane=half*16+(n%16), nt=n/16.
// A lane's whole 16-half fragment is then one contiguous 32-byte v16h.
__device__ __forceinline__ size_t fragIdx(int k, int n) {
  int kc = k >> 5, kk = k & 31;
  int lane = ((kk >> 4) << 4) | (n & 15);
  int nt = n >> 4;
  return (((size_t)((kc * 8 + nt) * 32 + lane)) << 4) | (unsigned)(kk & 15);
}

// ---------------------------------------------------------------------------
// Tensor Data Mover: 1-D contiguous f32 copy global -> LDS (ISA ch. 7/8).
// This toolchain exposes the 6-argument builtin:
//   (u32x4 group0, i32x8 group1, i32x4 group2, i32x4 group3, i32x8, i32 cpol)
// ---------------------------------------------------------------------------
#if defined(__gfx1250__) && __has_builtin(__builtin_amdgcn_tensor_load_to_lds) && \
    __has_builtin(__builtin_amdgcn_s_wait_tensorcnt)
#define HAVE_TDM 1
#endif

#ifdef HAVE_TDM
__device__ __forceinline__ void tdm_load_1d_f32(void* lds_dst, const void* gsrc,
                                                unsigned nelem /* f32 count */) {
  unsigned lds_off = (unsigned)(unsigned long long)(uintptr_t)lds_dst; // LDS byte offset
  unsigned long long ga = (unsigned long long)(uintptr_t)gsrc;
  u32x4 g0;
  g0[0] = 1u;                                                 // count=1, user descriptor
  g0[1] = lds_off;                                            // lds_addr
  g0[2] = (unsigned)(ga & 0xFFFFFFFFu);                       // global_addr[31:0]
  g0[3] = (unsigned)((ga >> 32) & 0x01FFFFFFu) | (2u << 30);  // global_addr[56:32], type=2
  i32x8 g1;
  g1[0] = (int)(2u << 16);                                    // data_size=4B; wg_mask=0
  g1[1] = (int)((nelem & 0xFFFFu) << 16);                     // tensor_dim0 lo
  g1[2] = (int)((nelem >> 16) & 0xFFFFu);                     // tensor_dim0 hi, dim1=0
  g1[3] = (int)((nelem & 0xFFFFu) << 16);                     // tile_dim0 = nelem (1-D)
  g1[4] = 0;                                                  // tile_dim1/2 = 0
  g1[5] = (int)nelem;                                         // tensor_dim0_stride lo
  g1[6] = 0;
  g1[7] = 0;
  i32x4 z4;
  z4[0] = 0; z4[1] = 0; z4[2] = 0; z4[3] = 0;
  i32x8 z8;
  z8[0] = 0; z8[1] = 0; z8[2] = 0; z8[3] = 0;
  z8[4] = 0; z8[5] = 0; z8[6] = 0; z8[7] = 0;
  __builtin_amdgcn_tensor_load_to_lds(g0, g1, z4, z4, z8, 0);
}
#endif

// ---------------------------------------------------------------------------
// Weight pre-swizzle: (K x 128) f32 -> fragment-linear f16.
// ---------------------------------------------------------------------------
__global__ void pack_w_kernel(const float* __restrict__ W, _Float16* __restrict__ Wf, int K) {
  int idx = blockIdx.x * blockDim.x + threadIdx.x;
  if (idx >= K * H_) return;
  int k = idx >> 7, n = idx & (H_ - 1);
  Wf[fragIdx(k, n)] = (_Float16)W[idx];
}

// ---------------------------------------------------------------------------
// Stage 1: h0 = relu(node_feats @ W_in + b_in); emit f32, f16 row-major, and
// fragment-linear f16 (B operand for the adjacency GEMM).
// ---------------------------------------------------------------------------
__global__ void embed_kernel(const float* __restrict__ feats,
                             const float* __restrict__ W_in,
                             const float* __restrict__ b_in,
                             float* __restrict__ h0,
                             _Float16* __restrict__ h016,
                             _Float16* __restrict__ h016f) {
  int idx = blockIdx.x * blockDim.x + threadIdx.x;
  if (idx >= B_ * N_ * H_) return;
  int n  = idx & (H_ - 1);
  int bn = idx >> 7;
  int b   = bn >> 9;          // N_ = 512
  int row = bn & (N_ - 1);
  const float* f = feats + (size_t)bn * FD_;
  float s = b_in[n];
#pragma unroll
  for (int k = 0; k < FD_; ++k) s += f[k] * W_in[k * H_ + n];
  s = s > 0.f ? s : 0.f;
  h0[idx]   = s;
  h016[idx] = (_Float16)s;
  h016f[(size_t)b * (N_ * H_) + fragIdx(row, n)] = (_Float16)s;
}

// ---------------------------------------------------------------------------
// Stage 2a: msg = adj @ h  (WMMA). No LDS, no barriers:
//  A: adj row pairs as float2 -> cvt;  B: fragment-linear h16f, v16h loads.
// block = 128 (4 waves, one 16-row i-tile each); grid = B*8.
// ---------------------------------------------------------------------------
__global__ __launch_bounds__(128)
void msg_kernel(const _Float16* __restrict__ h16f,  // [B][fragment-linear K=512]
                const float*    __restrict__ adj,   // [B,N,N]
                _Float16*       __restrict__ msg16) // [B,N,H] row-major
{
  int blk  = blockIdx.x;
  int b    = blk / 8;
  int grp  = blk % 8;
  int tid  = threadIdx.x;
  int wave = tid >> 5;
  int lane = tid & 31;
  int lq   = lane & 15;
  int half = lane >> 4;
  int i0   = (grp * 4 + wave) * 16;

  v8f z = {};
  v8f acc[8];
#pragma unroll
  for (int t = 0; t < 8; ++t) acc[t] = z;

  const float* arow = adj + ((size_t)b * N_ + (i0 + lq)) * N_;
  const v16h*  bf   = (const v16h*)(h16f + (size_t)b * (N_ * H_));

  for (int kc = 0; kc < N_ / 32; ++kc) {
    v16h afrag;
#pragma unroll
    for (int r = 0; r < 8; ++r) {
      int k = kc * 32 + kA16(r, half);
      float2 p = *(const float2*)(arow + k);
      afrag[2 * r]     = (_Float16)p.x;
      afrag[2 * r + 1] = (_Float16)p.y;
    }
#pragma unroll
    for (int nt = 0; nt < 8; ++nt) {
      v16h bfrag = bf[(kc * 8 + nt) * 32 + lane];
      acc[nt] = WMMA_F16(afrag, bfrag, acc[nt]);
    }
  }
  // C/D layout: lane holds (M = 8*half + v, n = nt*16 + lq)
#pragma unroll
  for (int nt = 0; nt < 8; ++nt)
#pragma unroll
    for (int v = 0; v < 8; ++v) {
      int row = i0 + 8 * half + v;
      msg16[((size_t)b * N_ + row) * H_ + nt * 16 + lq] = (_Float16)acc[nt][v];
    }
}

// ---------------------------------------------------------------------------
// Stage 2b: h' = relu([h || msg] @ W_mp[l] + b_mp[l])  (WMMA, K=256).
// A: packed u32 pair loads from row-major h16/msg16; B: fragment-linear Wmpf.
// ---------------------------------------------------------------------------
__global__ __launch_bounds__(128)
void mp_kernel(const _Float16* __restrict__ h16,    // [B,N,H] row-major
               const _Float16* __restrict__ msg16,  // [B,N,H] row-major
               const _Float16* __restrict__ Wmpf,   // fragment-linear [2H x H]
               const float*    __restrict__ bmp,    // [H]
               float*          __restrict__ h_out,
               _Float16*       __restrict__ h16_out,
               _Float16*       __restrict__ h16f_out) {
  int blk  = blockIdx.x;
  int b    = blk / 8;
  int grp  = blk % 8;
  int tid  = threadIdx.x;
  int wave = tid >> 5;
  int lane = tid & 31;
  int lq   = lane & 15;
  int half = lane >> 4;
  int i0   = (grp * 4 + wave) * 16;

  v8f z = {};
  v8f acc[8];
#pragma unroll
  for (int t = 0; t < 8; ++t) acc[t] = z;

  const unsigned* hrow32 = (const unsigned*)(h16   + ((size_t)b * N_ + (i0 + lq)) * H_);
  const unsigned* mrow32 = (const unsigned*)(msg16 + ((size_t)b * N_ + (i0 + lq)) * H_);
  const v16h* bf = (const v16h*)Wmpf;

  for (int kc = 0; kc < (2 * H_) / 32; ++kc) {
    v8u a8;
#pragma unroll
    for (int r = 0; r < 8; ++r) {
      int kg = kc * 32 + kA16(r, half);           // even; pair (kg, kg+1)
      a8[r] = (kg < H_) ? hrow32[kg >> 1] : mrow32[(kg - H_) >> 1];
    }
    v16h afrag = __builtin_bit_cast(v16h, a8);
#pragma unroll
    for (int nt = 0; nt < 8; ++nt) {
      v16h bfrag = bf[(kc * 8 + nt) * 32 + lane];
      acc[nt] = WMMA_F16(afrag, bfrag, acc[nt]);
    }
  }
#pragma unroll
  for (int nt = 0; nt < 8; ++nt) {
    float bb = bmp[nt * 16 + lq];
#pragma unroll
    for (int v = 0; v < 8; ++v) {
      float x = acc[nt][v] + bb;
      x = x > 0.f ? x : 0.f;
      int row  = i0 + 8 * half + v;
      int n    = nt * 16 + lq;
      size_t o = ((size_t)b * N_ + row) * H_ + n;
      h_out[o]    = x;
      h16_out[o]  = (_Float16)x;
      h16f_out[(size_t)b * (N_ * H_) + fragIdx(row, n)] = (_Float16)x;
    }
  }
}

// ---------------------------------------------------------------------------
// Stage 3: node heads -> type_full, state_full (small; VALU + LDS).
// ---------------------------------------------------------------------------
__global__ __launch_bounds__(128)
void heads_kernel(const float* __restrict__ h,
                  const float* __restrict__ feats,
                  const float* __restrict__ snm,  // [B,N]
                  const float* __restrict__ Wt1, const float* __restrict__ bt1,
                  const float* __restrict__ Wt2, const float* __restrict__ bt2,
                  const float* __restrict__ Ws1, const float* __restrict__ bs1,
                  const float* __restrict__ Ws2, const float* __restrict__ bs2,
                  float* __restrict__ type_out,   // [B,N,NT]
                  float* __restrict__ state_out)  // [B,N,SD]
{
  __shared__ float sh[H_], th[H_], sh1[H_];
  int bn  = blockIdx.x;
  int tid = threadIdx.x;
  sh[tid] = h[(size_t)bn * H_ + tid];
  __syncthreads();
  float t = bt1[tid], s = bs1[tid];
  for (int k = 0; k < H_; ++k) {
    float hv = sh[k];
    t += hv * Wt1[k * H_ + tid];
    s += hv * Ws1[k * H_ + tid];
  }
  th[tid]  = t > 0.f ? t : 0.f;
  sh1[tid] = s > 0.f ? s : 0.f;
  __syncthreads();
  float mask = snm[bn];
  if (tid < NT_) {
    float o = bt2[tid];
    for (int k = 0; k < H_; ++k) o += th[k] * Wt2[k * NT_ + tid];
    int cur = (int)feats[(size_t)bn * FD_];
    cur = cur < 0 ? 0 : (cur > NT_ - 1 ? NT_ - 1 : cur);
    float ct = (tid == cur) ? COPYV : -COPYV;
    type_out[(size_t)bn * NT_ + tid] = (mask != 0.f) ? o : ct;
  }
  if (tid >= 32 && tid < 32 + SD_) {
    int c = tid - 32;
    float o = bs2[c];
    for (int k = 0; k < H_; ++k) o += sh1[k] * Ws2[k * SD_ + c];
    state_out[(size_t)bn * SD_ + c] =
        mask * o + (1.f - mask) * feats[(size_t)bn * FD_ + 1 + c];
  }
}

// ---------------------------------------------------------------------------
// Stage 4: pairwise edge MLP (the 69-GFLOP WMMA workhorse).
// Block = 512 (16 waves) handles a (b, 16x16) (i,j) pair tile; wave w owns
// row i = i0+w. h tiles land in LDS via the Tensor Data Mover; A fragments
// synthesize [h_i | h_j | |h_i-h_j| | h_i*h_j] on the fly; B fragments are
// contiguous v16h loads from fragment-linear We1f. No barriers in the K loop.
// ---------------------------------------------------------------------------
__global__ __launch_bounds__(512)
void edge_mlp_kernel(const float*    __restrict__ h,     // [B,N,H] final
                     const _Float16* __restrict__ We1f,  // fragment-linear [4H x H]
                     const float*    __restrict__ be1,   // [H]
                     const float*    __restrict__ We2,   // [H,3]
                     const float*    __restrict__ be2,   // [3]
                     float* __restrict__ edl_pre)        // [B,N,N,3]
{
  __shared__ float s_hi[16][H_];  // 8 KB
  __shared__ float s_hj[16][H_];  // 8 KB

  int blk  = blockIdx.x;
  int b    = blk / (32 * 32);
  int it   = (blk / 32) % 32;
  int jt   = blk % 32;
  int i0   = it * 16, j0 = jt * 16;
  int tid  = threadIdx.x;
  int wave = tid >> 5;   // 0..15 -> row i0+wave
  int lane = tid & 31;
  int lq   = lane & 15;
  int half = lane >> 4;

#ifdef HAVE_TDM
  if (wave == 0) {
    // 16 consecutive node rows are one contiguous 8 KB span: a 1-D TDM tile.
    tdm_load_1d_f32(&s_hi[0][0], h + ((size_t)b * N_ + i0) * H_, 16 * H_);
    tdm_load_1d_f32(&s_hj[0][0], h + ((size_t)b * N_ + j0) * H_, 16 * H_);
    __builtin_amdgcn_s_wait_tensorcnt(0);
  }
  __syncthreads();
#else
  for (int e = tid; e < 16 * H_; e += 512) {
    int r = e >> 7, c = e & (H_ - 1);
    s_hi[r][c] = h[((size_t)b * N_ + (i0 + r)) * H_ + c];
    s_hj[r][c] = h[((size_t)b * N_ + (j0 + r)) * H_ + c];
  }
  __syncthreads();
#endif

  v8f z = {};
  v8f acc[8];
#pragma unroll
  for (int t = 0; t < 8; ++t) acc[t] = z;

  const v16h* bf = (const v16h*)We1f;

  // K = 512 = 4 feature sections x 4 chunks of 32; section select folds away.
  // NOTE: |x| is written as max(x, -x): the packed-fabs sign-mask idiom ICEs
  // this toolchain's gfx1250 ISel (v2f32 bitcast/and combine).
#pragma unroll
  for (int sec = 0; sec < 4; ++sec) {
    for (int kq = 0; kq < 4; ++kq) {
      int kc  = sec * 4 + kq;
      int kb0 = kq * 32;  // (kc*32) mod 128
      v16h afrag;
#pragma unroll
      for (int r = 0; r < 8; ++r) {
        int koff = kA16(r, half);
#pragma unroll
        for (int e2 = 0; e2 < 2; ++e2) {
          int kk = kb0 + koff + e2;
          float hi = s_hi[wave][kk];
          float hj = s_hj[lq][kk];
          float v = (sec == 0) ? hi
                  : (sec == 1) ? hj
                  : (sec == 2) ? fmaxf(hi - hj, hj - hi)
                               : hi * hj;
          afrag[2 * r + e2] = (_Float16)v;
        }
      }
#pragma unroll
      for (int nt = 0; nt < 8; ++nt) {
        v16h bfrag = bf[(kc * 8 + nt) * 32 + lane];
        acc[nt] = WMMA_F16(afrag, bfrag, acc[nt]);
      }
    }
  }

  // Epilogue: hid = relu(acc + be1); edl = hid @ We2 + be2 via width-16 butterflies.
  float hid[8][8];  // [nt][v]
#pragma unroll
  for (int nt = 0; nt < 8; ++nt) {
    float bb = be1[nt * 16 + lq];
#pragma unroll
    for (int v = 0; v < 8; ++v) {
      float x = acc[nt][v] + bb;
      hid[nt][v] = x > 0.f ? x : 0.f;
    }
  }
  float w2[8][3];
#pragma unroll
  for (int nt = 0; nt < 8; ++nt)
#pragma unroll
    for (int c = 0; c < 3; ++c) w2[nt][c] = We2[(nt * 16 + lq) * 3 + c];

  int i = i0 + wave;
#pragma unroll
  for (int v = 0; v < 8; ++v) {
    float o0 = 0.f, o1 = 0.f, o2 = 0.f;
#pragma unroll
    for (int nt = 0; nt < 8; ++nt) {
      float hv = hid[nt][v];
      o0 += hv * w2[nt][0];
      o1 += hv * w2[nt][1];
      o2 += hv * w2[nt][2];
    }
    for (int m = 1; m < 16; m <<= 1) {
      o0 += __shfl_xor(o0, m, 32);
      o1 += __shfl_xor(o1, m, 32);
      o2 += __shfl_xor(o2, m, 32);
    }
    if (lq == 0) {
      int j = j0 + 8 * half + v;
      size_t off = (((size_t)b * N_ + i) * N_ + j) * 3;
      edl_pre[off + 0] = o0 + be2[0];
      edl_pre[off + 1] = o1 + be2[1];
      edl_pre[off + 2] = o2 + be2[2];
    }
  }
}

// ---------------------------------------------------------------------------
// Stage 5: symmetrize, diag-zero, scope mixing, and edge-logit decode.
// ---------------------------------------------------------------------------
__global__ void finalize_kernel(const float* __restrict__ edl_pre,
                                const float* __restrict__ adj,
                                const float* __restrict__ sem,  // [B,N,N]
                                float* __restrict__ edl_full,   // [B,N,N,3]
                                float* __restrict__ el_full)    // [B,N,N]
{
  int idx = blockIdx.x * blockDim.x + threadIdx.x;
  if (idx >= B_ * N_ * N_) return;
  int j = idx % N_;
  int i = (idx / N_) % N_;
  int b = idx / (N_ * N_);
  size_t f  = (size_t)idx * 3;
  size_t ft = (((size_t)b * N_ + j) * N_ + i) * 3;

  bool diag = (i == j);
  float e0 = 0.f, e1 = 0.f, e2 = 0.f;
  if (!diag) {
    e0 = 0.5f * (edl_pre[f + 0] + edl_pre[ft + 0]);
    e1 = 0.5f * (edl_pre[f + 1] + edl_pre[ft + 1]);
    e2 = 0.5f * (edl_pre[f + 2] + edl_pre[ft + 2]);
  }
  float m  = sem[idx];
  float c0 = diag ? 0.f : COPYV;
  float c1 = diag ? 0.f : -COPYV;
  float d0 = (m != 0.f) ? e0 : c0;
  float d1 = (m != 0.f) ? e1 : c1;
  float d2 = (m != 0.f) ? e2 : c1;
  edl_full[f + 0] = d0;
  edl_full[f + 1] = d1;
  edl_full[f + 2] = d2;

  float mx  = fmaxf(d0, fmaxf(d1, d2));
  float x0  = __expf(d0 - mx), x1 = __expf(d1 - mx), x2 = __expf(d2 - mx);
  float inv = 1.f / (x0 + x1 + x2);
  float p_keep = x0 * inv, p_add = x1 * inv;
  float cur = (adj[idx] > 0.5f) ? 1.f : 0.f;
  float p = p_add + p_keep * cur;
  p = fminf(fmaxf(p, EPS_), 1.f - EPS_);
  float el = logf(p) - log1pf(-p);
  el_full[idx] = diag ? -1000000000.0f : el;
}

// ---------------------------------------------------------------------------
// Host launcher.
// Workspace layout (bytes), ~9 MB total:
//   hA f32 @0 (512K) | hB f32 @512K (512K)
//   h16A @1,048,576 (256K) | h16B @1,310,720 (256K) | msg16 @1,572,864 (256K)
//   h16fA @1,835,008 (256K) | h16fB @2,097,152 (256K)
//   We1f @2,359,296 (128K) | Wmpf @2,490,368 (192K)
//   edl_pre @2,686,976 (6,291,456)
// Output (float): type[3072] | state[4096] | edl_full[1572864] | el_full[524288]
// ---------------------------------------------------------------------------
extern "C" void kernel_launch(void* const* d_in, const int* in_sizes, int n_in,
                              void* d_out, int out_size, void* d_ws, size_t ws_size,
                              hipStream_t stream) {
  const float* feats = (const float*)d_in[0];
  const float* adj   = (const float*)d_in[1];
  const float* snm   = (const float*)d_in[2];
  const float* sem   = (const float*)d_in[3];
  const float* W_in  = (const float*)d_in[4];
  const float* b_in  = (const float*)d_in[5];
  const float* W_mp  = (const float*)d_in[6];
  const float* b_mp  = (const float*)d_in[7];
  const float* Wt1 = (const float*)d_in[8],  *bt1 = (const float*)d_in[9];
  const float* Wt2 = (const float*)d_in[10], *bt2 = (const float*)d_in[11];
  const float* Ws1 = (const float*)d_in[12], *bs1 = (const float*)d_in[13];
  const float* Ws2 = (const float*)d_in[14], *bs2 = (const float*)d_in[15];
  const float* We1 = (const float*)d_in[16], *be1 = (const float*)d_in[17];
  const float* We2 = (const float*)d_in[18], *be2 = (const float*)d_in[19];

  char* ws = (char*)d_ws;
  float*    hA      = (float*)(ws + 0);
  float*    hB      = (float*)(ws + 524288);
  _Float16* h16A    = (_Float16*)(ws + 1048576);
  _Float16* h16B    = (_Float16*)(ws + 1310720);
  _Float16* msg16   = (_Float16*)(ws + 1572864);
  _Float16* h16fA   = (_Float16*)(ws + 1835008);
  _Float16* h16fB   = (_Float16*)(ws + 2097152);
  _Float16* We1f    = (_Float16*)(ws + 2359296);
  _Float16* Wmpf    = (_Float16*)(ws + 2490368);
  float*    edl_pre = (float*)(ws + 2686976);

  float* type_out  = (float*)d_out;
  float* state_out = type_out + (size_t)B_ * N_ * NT_;
  float* edl_full  = state_out + (size_t)B_ * N_ * SD_;
  float* el_full   = edl_full + (size_t)B_ * N_ * N_ * 3;

  // Pre-swizzle weights into fragment-linear f16.
  pack_w_kernel<<<(4 * H_ * H_ + 255) / 256, 256, 0, stream>>>(We1, We1f, 4 * H_);
  for (int l = 0; l < 3; ++l)
    pack_w_kernel<<<(2 * H_ * H_ + 255) / 256, 256, 0, stream>>>(
        W_mp + (size_t)l * 2 * H_ * H_, Wmpf + (size_t)l * 2 * H_ * H_, 2 * H_);

  embed_kernel<<<(B_ * N_ * H_ + 255) / 256, 256, 0, stream>>>(feats, W_in, b_in,
                                                               hA, h16A, h16fA);

  const _Float16* hcur16     = h16A;
  const _Float16* hcur16f    = h16fA;
  float*          houts[3]   = {hB, hA, hB};
  _Float16*       hout16s[3] = {h16B, h16A, h16B};
  _Float16*       hout16fs[3]= {h16fB, h16fA, h16fB};
  for (int l = 0; l < 3; ++l) {
    msg_kernel<<<B_ * 8, 128, 0, stream>>>(hcur16f, adj, msg16);
    mp_kernel<<<B_ * 8, 128, 0, stream>>>(hcur16, msg16,
                                          Wmpf + (size_t)l * 2 * H_ * H_,
                                          b_mp + (size_t)l * H_,
                                          houts[l], hout16s[l], hout16fs[l]);
    hcur16  = hout16s[l];
    hcur16f = hout16fs[l];
  }
  const float* hfin = hB;  // after 3 layers

  heads_kernel<<<B_ * N_, 128, 0, stream>>>(hfin, feats, snm,
                                            Wt1, bt1, Wt2, bt2,
                                            Ws1, bs1, Ws2, bs2,
                                            type_out, state_out);

  edge_mlp_kernel<<<B_ * 32 * 32, 512, 0, stream>>>(hfin, We1f, be1, We2, be2, edl_pre);

  finalize_kernel<<<(B_ * N_ * N_ + 255) / 256, 256, 0, stream>>>(edl_pre, adj, sem,
                                                                  edl_full, el_full);
}
